// Attention_59184649339136
// MI455X (gfx1250) — compile-verified, matrix operand order checked by reading, not run on hardware
//
#include <hip/hip_runtime.h>
#include <hip/hip_bf16.h>
#include <stddef.h>
#include <stdint.h>

// Problem constants (match reference)
#define DIMC   1024
#define TDIMC  3072      // 3*DIM
#define BATCH  4
#define SEQ    2048
#define BT     8192      // BATCH*SEQ
#define NHEAD  16
#define HDIM   64

typedef __attribute__((ext_vector_type(8)))  __bf16 v8bf;
typedef __attribute__((ext_vector_type(16))) __bf16 v16bf;
typedef __attribute__((ext_vector_type(8)))  float  v8f;
typedef __attribute__((ext_vector_type(4)))  float  v4f;

__device__ __forceinline__ v8f wmma_bf16(v16bf a, v16bf b, v8f c) {
  return __builtin_amdgcn_wmma_f32_16x16x32_bf16(false, a, false, b,
                                                 (short)0, c, false, false);
}

__device__ __forceinline__ v16bf cat16(v8bf lo, v8bf hi) {
  return __builtin_shufflevector(lo, hi, 0, 1, 2, 3, 4, 5, 6, 7,
                                 8, 9, 10, 11, 12, 13, 14, 15);
}

// 16-bit WMMA operand: elements 0..7 <-> K = hi8+0..7 (16B contiguous),
// elements 8..15 <-> K = hi8+16..23 (16B contiguous).  hi8 = (lane>>4)*8.
__device__ __forceinline__ v16bf load_frag_g(const __bf16* rowp, int hi8) {
  return cat16(*(const v8bf*)(rowp + hi8), *(const v8bf*)(rowp + hi8 + 16));
}

// gfx1250 async global->LDS copy (ASYNCcnt path), 16 bytes per instruction.
__device__ __forceinline__ void async_b128(void* lds, const void* gaddr) {
  uint32_t l = (uint32_t)(uintptr_t)lds;       // low 32 bits = LDS byte address
  uint64_t g = (uint64_t)(uintptr_t)gaddr;
  asm volatile("global_load_async_to_lds_b128 %0, %1, off"
               :: "v"(l), "v"(g) : "memory");
}
#define ASYNC_WAIT(N) asm volatile("s_wait_asynccnt " #N ::: "memory")

// ---------------------------------------------------------------------------
// Prepass: f32 -> bf16 convert (x) and convert+transpose (weights, to [N][K])
// ---------------------------------------------------------------------------
__global__ void __launch_bounds__(256)
cvt_bf16_kernel(const float* __restrict__ src, __bf16* __restrict__ dst, int n) {
  const int i = blockIdx.x * 256 + threadIdx.x;
  if (i < n) dst[i] = (__bf16)src[i];
}

__global__ void __launch_bounds__(256)
transpose_bf16_kernel(const float* __restrict__ src, __bf16* __restrict__ dst,
                      int K, int N) {             // src [K][N] f32 -> dst [N][K] bf16
  const int i = blockIdx.x * 256 + threadIdx.x;
  if (i < K * N) {
    const int n = i / K, k = i % K;
    dst[i] = (__bf16)src[(size_t)k * N + n];
  }
}

// ---------------------------------------------------------------------------
// Kernel 1: QKV GEMM.  A = xbf [BT][K] bf16, B = wqkvT [N][K] bf16.
// Block tile 128x64: 8 waves x (16 rows x 64 cols).  B tile staged in LDS by
// async b128 copies, double buffered.  Writes Q (pre-scaled), K, V^T.
// ---------------------------------------------------------------------------
__global__ void __launch_bounds__(256)
qkv_gemm_kernel(const __bf16* __restrict__ xb, const __bf16* __restrict__ wT,
                __bf16* __restrict__ qws, __bf16* __restrict__ kws,
                __bf16* __restrict__ vtws)
{
  __shared__ __bf16 bbuf[2][64][32];              // [N-tile][K-step] 4KB each
  const int tid  = threadIdx.x;
  const int lane = tid & 31, wid = tid >> 5;
  const int NTB  = TDIMC / 64;                    // 48 col-tiles
  const int mt   = blockIdx.x / NTB;              // 128-row tile
  const int nt   = blockIdx.x % NTB;              // 64-col tile
  const int l16  = lane & 15, hi8 = (lane >> 4) << 3;
  const int arow = mt * 128 + wid * 16 + l16;
  const int cn   = tid >> 2;                      // copy row    (0..63)
  const int cc   = (tid & 3) * 8;                 // copy K-chunk (8 bf16 = 16B)
  const __bf16* wtile = wT + (size_t)(nt * 64 + cn) * DIMC + cc;

  v8f acc[4] = {};
  async_b128(&bbuf[0][cn][cc], wtile);
  for (int ks = 0; ks < DIMC / 32; ++ks) {
    const int k0 = ks * 32;
    if (ks + 1 < DIMC / 32) {
      async_b128(&bbuf[(ks + 1) & 1][cn][cc], wtile + (k0 + 32));
      ASYNC_WAIT(0x1);
    } else {
      ASYNC_WAIT(0x0);
    }
    __syncthreads();                              // buffer ks&1 ready everywhere

    const v16bf af = load_frag_g(&xb[(size_t)arow * DIMC + k0], hi8);
#pragma unroll
    for (int c = 0; c < 4; ++c) {
      const __bf16* brow = &bbuf[ks & 1][c * 16 + l16][0];
      acc[c] = wmma_bf16(af, cat16(*(const v8bf*)(brow + hi8),
                                   *(const v8bf*)(brow + hi8 + 16)), acc[c]);
    }
    __syncthreads();                              // done reading before overwrite
  }

  // Epilogue.  col = nt*64 + c*16 + l16; 64-col tile never crosses 1024.
  const int sel = nt / 16;                        // 0=q 1=k 2=v
  const int h   = nt % 16;
  const float scale = (sel == 0) ? 0.125f : 1.0f; // HEAD_DIM^-0.5 folded into Q
#pragma unroll
  for (int c = 0; c < 4; ++c) {
    const int d = c * 16 + l16;
#pragma unroll
    for (int r = 0; r < 8; ++r) {
      const int m = mt * 128 + wid * 16 + hi8 + r;    // row in [0, BT)
      const int b = m >> 11, t = m & (SEQ - 1);
      const __bf16 val = (__bf16)(acc[c][r] * scale);
      if (sel == 0)      qws[(((size_t)b * NHEAD + h) * SEQ + t) * HDIM + d] = val;
      else if (sel == 1) kws[(((size_t)b * NHEAD + h) * SEQ + t) * HDIM + d] = val;
      else               vtws[(((size_t)b * NHEAD + h) * HDIM + d) * SEQ + t] = val;
    }
  }
}

// ---------------------------------------------------------------------------
// Kernel 2: causal flash attention.  One wave = 16 query rows; block = 8 waves.
// All fragments fed by global_load_b128 pairs; P repacked via per-wave LDS.
// ---------------------------------------------------------------------------
__global__ void __launch_bounds__(256)
flash_attn_kernel(const __bf16* __restrict__ qws, const __bf16* __restrict__ kws,
                  const __bf16* __restrict__ vtws,
                  const unsigned char* __restrict__ amask,
                  __bf16* __restrict__ ows)
{
  __shared__ float plds[8][16 * 32];              // per-wave P tile (16KB)

  const int lane = threadIdx.x & 31;
  const int wid  = threadIdx.x >> 5;
  const int qt   = blockIdx.x & 15;
  const int bh   = blockIdx.x >> 4;
  const int b    = bh >> 4, h = bh & 15;
  const size_t base = (size_t)bh * SEQ * HDIM;
  const int qr0  = qt * 128 + wid * 16;
  const int l16  = lane & 15;
  const int hi8  = (lane >> 4) << 3;

  const __bf16* qrow = qws + base + (size_t)(qr0 + l16) * HDIM;
  v16bf qf[2] = { load_frag_g(qrow, hi8), load_frag_g(qrow + 32, hi8) };

  float mrow[8], lrow[8];
  v8f oacc[4] = {};
#pragma unroll
  for (int r = 0; r < 8; ++r) { mrow[r] = -1e30f; lrow[r] = 0.0f; }

  float* myl = plds[wid];
  const int nkb = (qr0 + 47) >> 5;

  for (int kb = 0; kb < nkb; ++kb) {
    const int s0 = kb * 32;

    v8f sc[2];
#pragma unroll
    for (int tt = 0; tt < 2; ++tt) {
      const int sBase = s0 + tt * 16;
      const __bf16* krow = kws + base + (size_t)(sBase + l16) * HDIM;
      v8f s = {};
      s = wmma_bf16(qf[0], load_frag_g(krow, hi8), s);
      s = wmma_bf16(qf[1], load_frag_g(krow + 32, hi8), s);

      const int key = sBase + l16;
      const bool kok = amask[(size_t)b * SEQ + key] != 0;
#pragma unroll
      for (int r = 0; r < 8; ++r) {
        const int row = qr0 + hi8 + r;
        if (key > row || !kok) s[r] = -1e30f;
      }
      sc[tt] = s;
    }

    float alpha[8];
#pragma unroll
    for (int r = 0; r < 8; ++r) {
      float mx = fmaxf(sc[0][r], sc[1][r]);
#pragma unroll
      for (int msk = 8; msk >= 1; msk >>= 1)
        mx = fmaxf(mx, __shfl_xor(mx, msk, 16));
      const float mnew = fmaxf(mrow[r], mx);
      alpha[r] = __expf(mrow[r] - mnew);
      mrow[r]  = mnew;
    }

#pragma unroll
    for (int tt = 0; tt < 2; ++tt)
#pragma unroll
      for (int r = 0; r < 8; ++r) {
        const float p = __expf(sc[tt][r] - mrow[r]);
        sc[tt][r] = p;
        myl[(hi8 + r) * 32 + tt * 16 + l16] = p;
      }
#pragma unroll
    for (int r = 0; r < 8; ++r) {
      float sum = sc[0][r] + sc[1][r];
#pragma unroll
      for (int msk = 8; msk >= 1; msk >>= 1)
        sum += __shfl_xor(sum, msk, 16);
      lrow[r] = lrow[r] * alpha[r] + sum;
    }

    // P (16x32 f32, row-major in LDS) -> bf16 A fragment: 4x ds_load_b128
    const float* prow = &myl[l16 * 32];
    const v4f p0 = *(const v4f*)(prow + hi8);
    const v4f p1 = *(const v4f*)(prow + hi8 + 4);
    const v4f p2 = *(const v4f*)(prow + hi8 + 16);
    const v4f p3 = *(const v4f*)(prow + hi8 + 20);
    v16bf pf;
#pragma unroll
    for (int i = 0; i < 4; ++i) {
      pf[i]      = (__bf16)p0[i];
      pf[4 + i]  = (__bf16)p1[i];
      pf[8 + i]  = (__bf16)p2[i];
      pf[12 + i] = (__bf16)p3[i];
    }

#pragma unroll
    for (int dt = 0; dt < 4; ++dt) {
#pragma unroll
      for (int r = 0; r < 8; ++r) oacc[dt][r] *= alpha[r];
      const __bf16* vrow = vtws + ((size_t)bh * HDIM + dt * 16 + l16) * SEQ + s0;
      oacc[dt] = wmma_bf16(pf, load_frag_g(vrow, hi8), oacc[dt]);
    }
  }

#pragma unroll
  for (int dt = 0; dt < 4; ++dt)
#pragma unroll
    for (int r = 0; r < 8; ++r) {
      const int row = qr0 + hi8 + r;
      const float inv = 1.0f / lrow[r];
      ows[((size_t)b * SEQ + row) * DIMC + h * HDIM + dt * 16 + l16] =
          (__bf16)(oacc[dt][r] * inv);
    }
}

// ---------------------------------------------------------------------------
// Kernel 3: out = O @ w_out.  Same structure as QKV GEMM; f32 output.
// ---------------------------------------------------------------------------
__global__ void __launch_bounds__(256)
out_gemm_kernel(const __bf16* __restrict__ a, const __bf16* __restrict__ wT,
                float* __restrict__ out)
{
  __shared__ __bf16 bbuf[2][64][32];
  const int tid  = threadIdx.x;
  const int lane = tid & 31, wid = tid >> 5;
  const int NTB  = DIMC / 64;                     // 16 col-tiles
  const int mt   = blockIdx.x / NTB;
  const int nt   = blockIdx.x % NTB;
  const int l16  = lane & 15, hi8 = (lane >> 4) << 3;
  const int arow = mt * 128 + wid * 16 + l16;
  const int cn   = tid >> 2;
  const int cc   = (tid & 3) * 8;
  const __bf16* wtile = wT + (size_t)(nt * 64 + cn) * DIMC + cc;

  v8f acc[4] = {};
  async_b128(&bbuf[0][cn][cc], wtile);
  for (int ks = 0; ks < DIMC / 32; ++ks) {
    const int k0 = ks * 32;
    if (ks + 1 < DIMC / 32) {
      async_b128(&bbuf[(ks + 1) & 1][cn][cc], wtile + (k0 + 32));
      ASYNC_WAIT(0x1);
    } else {
      ASYNC_WAIT(0x0);
    }
    __syncthreads();

    const v16bf af = load_frag_g(&a[(size_t)arow * DIMC + k0], hi8);
#pragma unroll
    for (int c = 0; c < 4; ++c) {
      const __bf16* brow = &bbuf[ks & 1][c * 16 + l16][0];
      acc[c] = wmma_bf16(af, cat16(*(const v8bf*)(brow + hi8),
                                   *(const v8bf*)(brow + hi8 + 16)), acc[c]);
    }
    __syncthreads();
  }

#pragma unroll
  for (int c = 0; c < 4; ++c)
#pragma unroll
    for (int r = 0; r < 8; ++r)
      out[(size_t)(mt * 128 + wid * 16 + hi8 + r) * DIMC + nt * 64 + c * 16 + l16] =
          acc[c][r];
}

// ---------------------------------------------------------------------------
// Launch
// ---------------------------------------------------------------------------
extern "C" void kernel_launch(void* const* d_in, const int* in_sizes, int n_in,
                              void* d_out, int out_size, void* d_ws, size_t ws_size,
                              hipStream_t stream) {
  (void)in_sizes; (void)n_in; (void)out_size; (void)ws_size;

  const float* x     = (const float*)d_in[0];            // [B,T,DIM] f32
  const float* w_qkv = (const float*)d_in[1];            // [DIM, 3*DIM] f32
  const float* w_out = (const float*)d_in[2];            // [DIM, DIM] f32
  const unsigned char* amask = (const unsigned char*)d_in[3];
  float* out = (float*)d_out;

  // Workspace (bf16): q, k, vT, o, xbf (BT*DIM each) + wqkvT + woutT  (~92 MB)
  __bf16* qws    = (__bf16*)d_ws;
  __bf16* kws    = qws    + (size_t)BT * DIMC;
  __bf16* vtws   = kws    + (size_t)BT * DIMC;
  __bf16* ows    = vtws   + (size_t)BT * DIMC;
  __bf16* xbf    = ows    + (size_t)BT * DIMC;
  __bf16* wqkvT  = xbf    + (size_t)BT * DIMC;
  __bf16* woutT  = wqkvT  + (size_t)DIMC * TDIMC;

  // 0) precision/layout prepass
  cvt_bf16_kernel<<<(BT * DIMC + 255) / 256, 256, 0, stream>>>(x, xbf, BT * DIMC);
  transpose_bf16_kernel<<<(DIMC * TDIMC + 255) / 256, 256, 0, stream>>>(
      w_qkv, wqkvT, DIMC, TDIMC);
  transpose_bf16_kernel<<<(DIMC * DIMC + 255) / 256, 256, 0, stream>>>(
      w_out, woutT, DIMC, DIMC);

  // 1) QKV projection: (BT/128) * (3DIM/64) = 64*48 = 3072 blocks
  qkv_gemm_kernel<<<3072, 256, 0, stream>>>(xbf, wqkvT, qws, kws, vtws);

  // 2) flash attention: B*H * (T/128) = 1024 blocks
  flash_attn_kernel<<<1024, 256, 0, stream>>>(qws, kws, vtws, amask, ows);

  // 3) output projection: (BT/128) * (DIM/64) = 64*16 = 1024 blocks
  out_gemm_kernel<<<1024, 256, 0, stream>>>(ows, woutT, out);
}
